// NeighbourhoodAttnBlock_82489141887274
// MI455X (gfx1250) — matrix operand order repaired and checked by reading, not compiled, not run on hardware
//
#include <hip/hip_runtime.h>
#include <hip/hip_bf16.h>

// ---------------------------------------------------------------------------
// Neighbourhood attention block (kernel=7) for B=2, H=W=64, D=384, 6 heads x 64
// bf16 WMMA (v_wmma_f32_16x16x32_bf16) for all matmuls, fp32 softmax.
// ---------------------------------------------------------------------------

typedef __bf16 bf16_t;
typedef __attribute__((ext_vector_type(16))) __bf16       v16bf;
typedef __attribute__((ext_vector_type(8)))  float        v8f;
typedef __attribute__((ext_vector_type(4)))  unsigned int u32x4;

#define BATCH  2
#define HH     64
#define WWID   64
#define DMODEL 384
#define NHEAD  6
#define DHEAD  64
#define NTOK   (HH * WWID)        // 4096 tokens per image
#define MROWS  (BATCH * NTOK)     // 8192
#define NQKV   (3 * DMODEL)       // 1152

union Frag {
  v16bf v;
  u32x4 q[2];
  unsigned int u[8];
};

__device__ __forceinline__ v8f wmma_bf16(v16bf a, v16bf b, v8f c) {
  // D = A(16x32) * B(32x16) + C, fp32 accumulate
  return __builtin_amdgcn_wmma_f32_16x16x32_bf16(false, a, false, b,
                                                 (short)0, c, false, false);
}

// A fragment from row-major [M,K] bf16. lane&15 selects M row (rowPtr is that
// row). ISA layout: lanes 0-15 hold K pairs {kk..kk+7, kk+16..kk+23},
// lanes 16-31 hold {kk+8..kk+15, kk+24..kk+31}. Two 16B loads.
__device__ __forceinline__ v16bf load_a_frag(const bf16_t* rowPtr, int kk, int lane) {
  const bf16_t* p = rowPtr + kk + ((lane >> 4) & 1) * 8;
  Frag f;
  f.q[0] = *(const u32x4*)(p);
  f.q[1] = *(const u32x4*)(p + 16);
  return f.v;
}

// B fragment: lane = K row, 16 contiguous bf16 (N columns), 32B-aligned.
__device__ __forceinline__ v16bf load_b_frag_al(const bf16_t* p) {
  Frag f;
  f.q[0] = *(const u32x4*)(p);
  f.q[1] = *(const u32x4*)(p + 8);
  return f.v;
}

// Same, but only 4B alignment guaranteed (dword loads).
__device__ __forceinline__ v16bf load_b_frag_u(const bf16_t* p) {
  Frag f;
#pragma unroll
  for (int i = 0; i < 8; ++i) f.u[i] = *(const unsigned int*)(p + 2 * i);
  return f.v;
}

// ---------------------------------------------------------------------------
// Phase 1: fp32 -> bf16 conversions (+ weight transposes to K-major)
// ---------------------------------------------------------------------------
__global__ __launch_bounds__(256)
void na_convert_kernel(const float* __restrict__ x, const float* __restrict__ wqkv,
                       const float* __restrict__ wout, bf16_t* __restrict__ Xb,
                       bf16_t* __restrict__ WqT, bf16_t* __restrict__ WoT) {
  const int NX  = MROWS * DMODEL;   // 3145728
  const int NWQ = NQKV * DMODEL;    // 442368
  const int NWO = DMODEL * DMODEL;  // 147456
  int i = blockIdx.x * blockDim.x + threadIdx.x;
  if (i < NX) {
    Xb[i] = (bf16_t)x[i];
  } else if (i < NX + NWQ) {
    int j = i - NX;
    int n = j / DMODEL, k = j - n * DMODEL;
    WqT[n * DMODEL + k] = (bf16_t)wqkv[j];     // [N=1152][K=384]
  } else if (i < NX + NWQ + NWO) {
    int j = i - NX - NWQ;
    int n = j / DMODEL, k = j - n * DMODEL;
    WoT[n * DMODEL + k] = (bf16_t)wout[j];     // [N=384][K=384]
  }
}

// ---------------------------------------------------------------------------
// Phase 2: QKV GEMM  [8192,384] x [384,1152] -> split into Q/K/V buffers
//   Q,K,V: [bh][token][64] row-major
// One wave computes a 16x64 output tile.
// ---------------------------------------------------------------------------
__global__ __launch_bounds__(128)
void na_qkv_gemm_kernel(const bf16_t* __restrict__ Xb, const bf16_t* __restrict__ WqT,
                        bf16_t* __restrict__ qb, bf16_t* __restrict__ kb,
                        bf16_t* __restrict__ vb) {
  int wave = (blockIdx.x * blockDim.x + threadIdx.x) >> 5;
  int lane = threadIdx.x & 31;
  const int NG = NQKV / 64;  // 18 column groups
  int mt = wave / NG;
  int ng = wave - mt * NG;
  if (mt >= MROWS / 16) return;
  int n0 = ng * 64;

  v8f acc[4] = {};
  const bf16_t* aRow = Xb + (size_t)(mt * 16 + (lane & 15)) * DMODEL;
#pragma unroll 4
  for (int kk = 0; kk < DMODEL; kk += 32) {
    v16bf a = load_a_frag(aRow, kk, lane);
    const bf16_t* bp = WqT + (size_t)(n0 + (lane & 15)) * DMODEL;
#pragma unroll
    for (int nt = 0; nt < 4; ++nt)
      acc[nt] = wmma_bf16(a, load_a_frag(bp + (size_t)nt * 16 * DMODEL, kk, lane), acc[nt]);
  }

  int col = lane & 15, hi = lane >> 4;
#pragma unroll
  for (int nt = 0; nt < 4; ++nt) {
    int r = n0 + nt * 16 + col;             // 0..1151 (t,head uniform per nt)
    int t = r / DMODEL;
    int rem = r - t * DMODEL;
    int head = rem >> 6;
    int e = rem & 63;
#pragma unroll
    for (int j = 0; j < 8; ++j) {
      int token = mt * 16 + j + 8 * hi;     // 0..8191
      int b = token >> 12;
      int n = token & (NTOK - 1);
      int bh = b * NHEAD + head;
      bf16_t val = (bf16_t)acc[nt][j];
      if (t == 0)      qb[(((size_t)bh * NTOK + n) << 6) + e] = val;
      else if (t == 1) kb[(((size_t)bh * NTOK + n) << 6) + e] = val;
      else             vb[(((size_t)bh * NTOK + n) << 6) + e] = val;
    }
  }
}

// ---------------------------------------------------------------------------
// Phase 3: neighbourhood attention. One wave per (b, head, 2x8 query tile).
// Key region: 8 rows x 16 cols (always covers all 7x7 windows of the tile).
// S = Q K^T (16 WMMA) -> mask -> softmax -> P (via LDS) -> P V (16 WMMA).
// ---------------------------------------------------------------------------
__global__ __launch_bounds__(32)
void na_attn_kernel(const bf16_t* __restrict__ qb, const bf16_t* __restrict__ kb,
                    const bf16_t* __restrict__ vb, bf16_t* __restrict__ attn) {
  __shared__ __align__(16) bf16_t sP[16 * 136];   // 16 rows x 128 keys, padded

  int lane = threadIdx.x;
  int bid  = blockIdx.x;            // 0..3071
  int bh   = bid >> 8;              // 0..11  (b*6+head)
  int tile = bid & 255;
  int qh0  = (tile >> 3) * 2;       // query block row (2 rows)
  int qw0  = (tile & 7) * 8;        // query block col (8 cols)
  int b    = bh / NHEAD;
  int head = bh - b * NHEAD;
  int kr0  = min(max(qh0 - 3, 0), 56);                 // key region top
  int kc0  = (qw0 == 0) ? 0 : min(qw0 - 4, 48);        // key region left (even)
  int col = lane & 15, hi = lane >> 4;

  // ---- S = Q K^T -------------------------------------------------------
  int am   = col;                                       // A lane -> query m
  int aTok = (qh0 + (am >> 3)) * WWID + qw0 + (am & 7);
  const bf16_t* qRow  = qb + ((size_t)(bh * NTOK + aTok) << 6);
  const bf16_t* kBase = kb + ((size_t)bh << 6) * NTOK;

  v8f sAcc[8] = {};
#pragma unroll
  for (int kkh = 0; kkh < 2; ++kkh) {
    int kk = kkh * 32;
    v16bf aQ = load_a_frag(qRow, kk, lane);
    const bf16_t* kp = kBase + ((size_t)(kr0 * WWID + kc0 + col) << 6);
#pragma unroll
    for (int jt = 0; jt < 8; ++jt)                      // key region row jt
      sAcc[jt] = wmma_bf16(aQ, load_a_frag(kp + (size_t)jt * WWID * 64, kk, lane), sAcc[jt]);
  }

  // ---- mask + softmax (rows distributed per C-layout) ------------------
  int qh = qh0 + hi;                       // this lane's query row
  int sh = min(max(qh - 3, 0), 57);
  int kw = kc0 + col;                      // this lane's key column
  float rinv[8];
#pragma unroll
  for (int v = 0; v < 8; ++v) {            // query col qw0+v, row m = v+8*hi
    int qw = qw0 + v;
    int sw = min(max(qw - 3, 0), 57);
    bool wok = (kw >= sw) && (kw < sw + 7);
    float m = -1e30f;
#pragma unroll
    for (int jt = 0; jt < 8; ++jt) {
      int kh = kr0 + jt;
      bool ok = wok && (kh >= sh) && (kh < sh + 7);
      float s = ok ? sAcc[jt][v] * 0.125f : -1e30f;     // 1/sqrt(64)
      sAcc[jt][v] = s;
      m = fmaxf(m, s);
    }
#pragma unroll
    for (int off = 1; off < 16; off <<= 1) m = fmaxf(m, __shfl_xor(m, off, 16));
    float sum = 0.f;
#pragma unroll
    for (int jt = 0; jt < 8; ++jt) {
      float p = __expf(sAcc[jt][v] - m);
      sAcc[jt][v] = p;
      sum += p;
    }
#pragma unroll
    for (int off = 1; off < 16; off <<= 1) sum += __shfl_xor(sum, off, 16);
    rinv[v] = 1.0f / sum;                  // >= 49 valid keys, never zero
  }

  // ---- C-layout -> A-layout through LDS --------------------------------
#pragma unroll
  for (int jt = 0; jt < 8; ++jt)
#pragma unroll
    for (int v = 0; v < 8; ++v)
      sP[(v + 8 * hi) * 136 + jt * 16 + col] = (bf16_t)sAcc[jt][v];
  __syncthreads();

  // ---- O = P V ---------------------------------------------------------
  v8f oAcc[4] = {};
  const bf16_t* vBase = vb + ((size_t)(bh * NTOK) << 6);
#pragma unroll
  for (int kkt = 0; kkt < 4; ++kkt) {
    int kk = kkt * 32;
    const bf16_t* pp = sP + col * 136 + kk + hi * 8;
    Frag fa;
    fa.q[0] = *(const u32x4*)(pp);
    fa.q[1] = *(const u32x4*)(pp + 16);
#pragma unroll
    for (int nt = 0; nt < 4; ++nt) {
      Frag fb;
#pragma unroll
      for (int s = 0; s < 16; ++s) {
        int rk  = kk + ((s >> 3) << 4) + hi * 8 + (s & 7);
        int tok = (kr0 + (rk >> 4)) * WWID + kc0 + (rk & 15);
        fb.v[s] = vBase[((size_t)tok << 6) + nt * 16 + col];
      }
      oAcc[nt] = wmma_bf16(fa.v, fb.v, oAcc[nt]);
    }
  }

  // ---- normalize + store [b][token][head*64+e] bf16 --------------------
#pragma unroll
  for (int nt = 0; nt < 4; ++nt) {
    int e = nt * 16 + col;
#pragma unroll
    for (int v = 0; v < 8; ++v) {
      int qtok = (qh0 + hi) * WWID + qw0 + v;
      float val = oAcc[nt][v] * rinv[v];
      attn[((size_t)(b * NTOK + qtok)) * DMODEL + head * 64 + e] = (bf16_t)val;
    }
  }
}

// ---------------------------------------------------------------------------
// Phase 4: output GEMM  [8192,384] x [384,384] -> fp32 d_out
// ---------------------------------------------------------------------------
__global__ __launch_bounds__(128)
void na_out_gemm_kernel(const bf16_t* __restrict__ Ab, const bf16_t* __restrict__ WoT,
                        float* __restrict__ C) {
  int wave = (blockIdx.x * blockDim.x + threadIdx.x) >> 5;
  int lane = threadIdx.x & 31;
  const int NG = DMODEL / 64;  // 6
  int mt = wave / NG;
  int ng = wave - mt * NG;
  if (mt >= MROWS / 16) return;
  int n0 = ng * 64;

  v8f acc[4] = {};
  const bf16_t* aRow = Ab + (size_t)(mt * 16 + (lane & 15)) * DMODEL;
#pragma unroll 4
  for (int kk = 0; kk < DMODEL; kk += 32) {
    v16bf a = load_a_frag(aRow, kk, lane);
    const bf16_t* bp = WoT + (size_t)(n0 + (lane & 15)) * DMODEL;
#pragma unroll
    for (int nt = 0; nt < 4; ++nt)
      acc[nt] = wmma_bf16(a, load_a_frag(bp + (size_t)nt * 16 * DMODEL, kk, lane), acc[nt]);
  }

  int col = lane & 15, hi = lane >> 4;
#pragma unroll
  for (int nt = 0; nt < 4; ++nt)
#pragma unroll
    for (int j = 0; j < 8; ++j)
      C[(size_t)(mt * 16 + j + 8 * hi) * DMODEL + n0 + nt * 16 + col] = acc[nt][j];
}

// ---------------------------------------------------------------------------
extern "C" void kernel_launch(void* const* d_in, const int* in_sizes, int n_in,
                              void* d_out, int out_size, void* d_ws, size_t ws_size,
                              hipStream_t stream) {
  const float* x    = (const float*)d_in[0];
  const float* wqkv = (const float*)d_in[1];
  const float* wout = (const float*)d_in[2];
  float* out = (float*)d_out;
  char* ws = (char*)d_ws;

  size_t off = 0;
  bf16_t* Xb   = (bf16_t*)(ws + off); off += (size_t)MROWS * DMODEL * 2;      // 6.29 MB
  bf16_t* WqT  = (bf16_t*)(ws + off); off += (size_t)DMODEL * NQKV * 2;       // 0.88 MB
  bf16_t* WoT  = (bf16_t*)(ws + off); off += (size_t)DMODEL * DMODEL * 2;     // 0.29 MB
  bf16_t* qbuf = (bf16_t*)(ws + off); off += (size_t)BATCH * NHEAD * NTOK * DHEAD * 2;
  bf16_t* kbuf = (bf16_t*)(ws + off); off += (size_t)BATCH * NHEAD * NTOK * DHEAD * 2;
  bf16_t* vbuf = (bf16_t*)(ws + off); off += (size_t)BATCH * NHEAD * NTOK * DHEAD * 2;
  bf16_t* attn = (bf16_t*)(ws + off); off += (size_t)MROWS * DMODEL * 2;      // total ~31 MB

  int totalConv = MROWS * DMODEL + NQKV * DMODEL + DMODEL * DMODEL;
  na_convert_kernel<<<(totalConv + 255) / 256, 256, 0, stream>>>(x, wqkv, wout,
                                                                 Xb, WqT, WoT);
  na_qkv_gemm_kernel<<<(MROWS / 16) * (NQKV / 64) / 4, 128, 0, stream>>>(
      Xb, WqT, qbuf, kbuf, vbuf);
  na_attn_kernel<<<BATCH * NHEAD * (HH / 2) * (WWID / 8), 32, 0, stream>>>(
      qbuf, kbuf, vbuf, attn);
  na_out_gemm_kernel<<<(MROWS / 16) * (DMODEL / 64) / 4, 128, 0, stream>>>(
      attn, WoT, out);
}
